// MDGCRNAdjHiD_89928025244556
// MI455X (gfx1250) — compile-verified, matrix operand order checked by reading, not run on hardware
//
#include <hip/hip_runtime.h>
#include <hip/hip_bf16.h>

// ---------------------------------------------------------------------------
// Model constants (match reference)
// ---------------------------------------------------------------------------
#define BB       64
#define TT       12
#define NN       512
#define HORIZON_ 12
#define RNN_     64
#define DEC_     128     // RNN + MEM_DIM
#define MEMN_    20
#define MEMD_    64
#define EMB_     10
#define ENC_IN   65      // IN_DIM + RNN          (real feature width, encoder)
#define DEC_IN   130     // OUT + YCOV + DEC_DIM  (real feature width, decoder)
#define ENC_CP   80      // ENC_IN padded to multiple of 16
#define DEC_CP   144     // DEC_IN padded to multiple of 16
#define ENC_3C   195     // CHEB_K * ENC_IN (real)
#define DEC_3C   390     // CHEB_K * DEC_IN (real)
#define ENC_KP   224     // ENC_3C padded to multiple of 32
#define DEC_KP   416     // DEC_3C padded to multiple of 32

#define CDIV(a, b) (((a) + (b) - 1) / (b))

typedef __attribute__((ext_vector_type(16))) _Float16 v16h;
typedef __attribute__((ext_vector_type(8)))  _Float16 v8h;
typedef __attribute__((ext_vector_type(8)))  float    v8f;

// ---------------------------------------------------------------------------
// Batched WMMA GEMM:  CT = act( A[M,K] @ B[K,Nn] + bias[Nn] )^T
//   A  : f16 row-major [M][K]            (M%16==0, K%32==0)
//   BT : f16 row-major [Nn][K]           (B transposed; Nn%16==0)
//   CT : f32 row-major [Nn][M]           (C transposed -> contiguous stores)
//   CTh: optional f16 shadow of CT (feeds next GEMM's BT directly)
// One wave computes one 16x16 tile with v_wmma_f32_16x16x32_f16.
// No bounds checks: caller guarantees padded dimensions. All fragment loads
// are contiguous 16-byte (b128) loads.
// act: 0 = none, 1 = sigmoid, 2 = tanh
// ---------------------------------------------------------------------------
__global__ __launch_bounds__(256)
void gemm_wmma_f16(const _Float16* __restrict__ A,  long long sA,
                   const _Float16* __restrict__ BT, long long sB,
                   float* __restrict__ CT,          long long sC,
                   _Float16* __restrict__ CTh,      long long sCh,
                   const float* __restrict__ bias,
                   int M, int Nn, int K, int act)
{
    const int batch = blockIdx.y;
    const _Float16* Ab = A  + (long long)batch * sA;
    const _Float16* Bb = BT + (long long)batch * sB;
    float*          Cb = CT + (long long)batch * sC;

    const int tilesN = Nn >> 4;
    const int tilesM = M  >> 4;
    const int wid    = threadIdx.x >> 5;
    const int lane   = threadIdx.x & 31;
    const int tile   = blockIdx.x * 8 + wid;
    if (tile >= tilesM * tilesN) return;
    const int tm = tile / tilesN;
    const int tn = tile % tilesN;

    // Fragment addressing (16-bit 16x32 layout, A and B mirrored):
    //   lanes 0-15 : elems 0-7 -> K = k0+0..7   ; elems 8-15 -> K = k0+16..23
    //   lanes 16-31: elems 0-7 -> K = k0+8..15  ; elems 8-15 -> K = k0+24..31
    const int r    = lane & 15;
    const int koff = (lane >> 4) << 3;            // 0 or 8

    const _Float16* ap = Ab + (long long)(tm * 16 + r) * K + koff;
    const _Float16* bp = Bb + (long long)(tn * 16 + r) * K + koff;

    v8f acc = {};
#pragma unroll 4
    for (int k0 = 0; k0 < K; k0 += 32) {
        __builtin_prefetch(ap + 256, 0, 3);
        const v8h a0 = *(const v8h*)(ap);
        const v8h a1 = *(const v8h*)(ap + 16);
        const v8h b0 = *(const v8h*)(bp);
        const v8h b1 = *(const v8h*)(bp + 16);
        const v16h a = __builtin_shufflevector(a0, a1, 0, 1, 2, 3, 4, 5, 6, 7,
                                               8, 9, 10, 11, 12, 13, 14, 15);
        const v16h b = __builtin_shufflevector(b0, b1, 0, 1, 2, 3, 4, 5, 6, 7,
                                               8, 9, 10, 11, 12, 13, 14, 15);
        acc = __builtin_amdgcn_wmma_f32_16x16x32_f16(
                  /*neg_a=*/false, a, /*neg_b=*/false, b,
                  /*c_mod=*/(short)0, acc, /*reuse_a=*/false, /*reuse_b=*/false);
        ap += 32;
        bp += 32;
    }

    // Epilogue: lane holds 8 consecutive M values of column `col`.
    const int col   = tn * 16 + r;
    const int mbase = tm * 16 + ((lane >> 4) << 3);
    const float bv  = bias ? bias[col] : 0.f;

    float*    cp  = Cb + (long long)col * M + mbase;
    _Float16* chp = CTh ? (CTh + (long long)batch * sCh + (long long)col * M + mbase)
                        : (_Float16*)0;
#pragma unroll
    for (int v = 0; v < 8; ++v) {
        float val = acc[v] + bv;
        if (act == 1)      val = 1.f / (1.f + __expf(-val));
        else if (act == 2) val = tanhf(val);
        cp[v] = val;
        if (chp) chp[v] = (_Float16)val;
    }
}

// ---------------------------------------------------------------------------
// Small helper kernels
// ---------------------------------------------------------------------------
__global__ void k_zero_f32(float* p, long long n) {
    long long i = (long long)blockIdx.x * blockDim.x + threadIdx.x;
    if (i < n) p[i] = 0.f;
}

__global__ void k_f32_to_f16(const float* __restrict__ s, _Float16* __restrict__ d, long long n) {
    long long i = (long long)blockIdx.x * blockDim.x + threadIdx.x;
    if (i < n) d[i] = (_Float16)s[i];
}

// WT[nf][k] = W[k][nf] for k < Kreal, else 0 (zero pads so garbage in the
// padded K columns of the activation operand contributes nothing).
__global__ void k_w_transpose_f16(const float* __restrict__ W, _Float16* __restrict__ WT,
                                  int Kreal, int KP, int Nf) {
    long long i = (long long)blockIdx.x * blockDim.x + threadIdx.x;
    if (i >= (long long)Nf * KP) return;
    const int k  = (int)(i % KP);
    const int nf = (int)(i / KP);
    WT[i] = (k < Kreal) ? (_Float16)W[(long long)k * Nf + nf] : (_Float16)0.f;
}

// Encoder graph operand, transposed layout: inpT[b][c][n] ; c<ENC_IN real.
__global__ void k_enc_inp(const float* __restrict__ x, const float* __restrict__ h,
                          _Float16* __restrict__ inpT, int t) {
    long long i = (long long)blockIdx.x * blockDim.x + threadIdx.x;
    if (i >= (long long)BB * ENC_IN * NN) return;
    const int n = (int)(i % NN);
    const long long rem = i / NN;
    const int c = (int)(rem % ENC_IN);
    const int b = (int)(rem / ENC_IN);
    const float v = (c == 0) ? x[((long long)b * TT + t) * NN + n]
                             : h[((long long)b * NN + n) * RNN_ + (c - 1)];
    inpT[((long long)b * ENC_CP + c) * NN + n] = (_Float16)v;
}

// Encoder candidate operand: [x, z*h] ; z = zrT[j][bn]
__global__ void k_enc_cand(const float* __restrict__ x, const float* __restrict__ zrT,
                           const float* __restrict__ h, _Float16* __restrict__ inpT, int t) {
    long long i = (long long)blockIdx.x * blockDim.x + threadIdx.x;
    if (i >= (long long)BB * ENC_IN * NN) return;
    const int n = (int)(i % NN);
    const long long rem = i / NN;
    const int c = (int)(rem % ENC_IN);
    const int b = (int)(rem / ENC_IN);
    const long long bn = (long long)b * NN + n;
    float v;
    if (c == 0) v = x[((long long)b * TT + t) * NN + n];
    else {
        const int j = c - 1;
        v = zrT[(long long)j * (BB * NN) + bn] * h[bn * RNN_ + j];
    }
    inpT[((long long)b * ENC_CP + c) * NN + n] = (_Float16)v;
}

// Decoder graph operand: [go, ycov, h] transposed
__global__ void k_dec_inp(const float* __restrict__ go, const float* __restrict__ yc,
                          const float* __restrict__ h, _Float16* __restrict__ inpT, int t) {
    long long i = (long long)blockIdx.x * blockDim.x + threadIdx.x;
    if (i >= (long long)BB * DEC_IN * NN) return;
    const int n = (int)(i % NN);
    const long long rem = i / NN;
    const int c = (int)(rem % DEC_IN);
    const int b = (int)(rem / DEC_IN);
    const long long bn = (long long)b * NN + n;
    float v;
    if (c == 0)      v = go[bn];
    else if (c == 1) v = yc[((long long)b * HORIZON_ + t) * NN + n];
    else             v = h[bn * DEC_ + (c - 2)];
    inpT[((long long)b * DEC_CP + c) * NN + n] = (_Float16)v;
}

__global__ void k_dec_cand(const float* __restrict__ go, const float* __restrict__ yc,
                           const float* __restrict__ zrT, const float* __restrict__ h,
                           _Float16* __restrict__ inpT, int t) {
    long long i = (long long)blockIdx.x * blockDim.x + threadIdx.x;
    if (i >= (long long)BB * DEC_IN * NN) return;
    const int n = (int)(i % NN);
    const long long rem = i / NN;
    const int c = (int)(rem % DEC_IN);
    const int b = (int)(rem / DEC_IN);
    const long long bn = (long long)b * NN + n;
    float v;
    if (c == 0)      v = go[bn];
    else if (c == 1) v = yc[((long long)b * HORIZON_ + t) * NN + n];
    else {
        const int j = c - 2;
        v = zrT[(long long)j * (BB * NN) + bn] * h[bn * DEC_ + j];
    }
    inpT[((long long)b * DEC_CP + c) * NN + n] = (_Float16)v;
}

// Chebyshev concat into weight-GEMM A operand (row-major [bn][KP]):
//   cheb[bn][c] = x0 ; cheb[bn][C+c] = g1 ; cheb[bn][2C+c] = 2*g2 - x0
// Reads the transposed graph-GEMM outputs.
__global__ void k_cheb_concat(const _Float16* __restrict__ inpT, const float* __restrict__ g1T,
                              const float* __restrict__ g2T, _Float16* __restrict__ cheb,
                              int C, int CP, int KP) {
    long long i = (long long)blockIdx.x * blockDim.x + threadIdx.x;
    if (i >= (long long)BB * C * NN) return;
    const int n = (int)(i % NN);
    const long long rem = i / NN;
    const int c = (int)(rem % C);
    const int b = (int)(rem / C);
    const long long it = ((long long)b * CP + c) * NN + n;
    const float x0  = (float)inpT[it];
    const float g1v = g1T[it];
    const float g2v = g2T[it];
    _Float16* o = cheb + ((long long)b * NN + n) * KP;
    o[c]         = (_Float16)x0;
    o[C + c]     = (_Float16)g1v;
    o[2 * C + c] = (_Float16)(2.f * g2v - x0);
}

// h = r*h + (1-r)*hc ; transposed gate/candidate layouts
__global__ void k_gru_update(const float* __restrict__ zrT, const float* __restrict__ hcT,
                             float* __restrict__ h, int Hd) {
    long long i = (long long)blockIdx.x * blockDim.x + threadIdx.x;
    const long long BN = (long long)BB * NN;
    if (i >= BN * Hd) return;
    const long long bn = i % BN;
    const int j = (int)(i / BN);
    const float r  = zrT[((long long)(Hd + j)) * BN + bn];
    const float hv = hcT[(long long)j * BN + bn];
    float* hp = h + bn * Hd + j;
    *hp = r * (*hp) + (1.f - r) * hv;
}

// Per-node memory attention: q = h@Wq ; att = softmax(q@Mem^T) ; value = att@Mem
__global__ void k_memory_query(const float* __restrict__ h, const float* __restrict__ Mem,
                               const float* __restrict__ Wq, float* __restrict__ hAtt,
                               int* __restrict__ ind) {
    const long long i = (long long)blockIdx.x * blockDim.x + threadIdx.x;
    if (i >= (long long)BB * NN) return;
    const float* hr = h + i * RNN_;
    float q[MEMD_];
    for (int j = 0; j < MEMD_; ++j) {
        float s = 0.f;
        for (int k = 0; k < RNN_; ++k) s += hr[k] * Wq[k * MEMD_ + j];
        q[j] = s;
    }
    float sc[MEMN_], mx = -1e30f;
    for (int m = 0; m < MEMN_; ++m) {
        float s = 0.f;
        for (int k = 0; k < MEMD_; ++k) s += q[k] * Mem[m * MEMD_ + k];
        sc[m] = s; mx = fmaxf(mx, s);
    }
    float sum = 0.f;
    for (int m = 0; m < MEMN_; ++m) { sc[m] = __expf(sc[m] - mx); sum += sc[m]; }
    const float inv = 1.f / sum;
    int best = 0; float bv = -1.f;
    for (int m = 0; m < MEMN_; ++m) {
        sc[m] *= inv;
        if (sc[m] > bv) { bv = sc[m]; best = m; }
    }
    ind[i] = best;
    if (hAtt) {
        for (int j = 0; j < MEMD_; ++j) {
            float s = 0.f;
            for (int m = 0; m < MEMN_; ++m) s += sc[m] * Mem[m * MEMD_ + j];
            hAtt[i * MEMD_ + j] = s;
        }
    }
}

__global__ void k_latent(const float* __restrict__ Mem, const int* __restrict__ i0,
                         const int* __restrict__ i1, const float* __restrict__ lw,
                         const float* __restrict__ lb, float* __restrict__ out) {
    const long long i = (long long)blockIdx.x * blockDim.x + threadIdx.x;
    if (i >= (long long)BB * NN) return;
    const float* a = Mem + (long long)i0[i] * MEMD_;
    const float* b = Mem + (long long)i1[i] * MEMD_;
    float s = lb[0];
    for (int k = 0; k < MEMD_; ++k) s += (a[k] - b[k]) * lw[k];
    out[i] = 1.f / (1.f + __expf(-s));
}

// hAug = [h, hAtt] ; emb = hAug @ hyperW + hyperb
__global__ void k_haug_emb(const float* __restrict__ h, const float* __restrict__ hAtt,
                           float* __restrict__ hAug, float* __restrict__ emb,
                           const float* __restrict__ hW, const float* __restrict__ hb) {
    const long long i = (long long)blockIdx.x * blockDim.x + threadIdx.x;
    if (i >= (long long)BB * NN) return;
    float loc[DEC_];
    for (int j = 0; j < RNN_; ++j) {
        loc[j]        = h[i * RNN_ + j];
        loc[RNN_ + j] = hAtt[i * MEMD_ + j];
    }
    for (int j = 0; j < DEC_; ++j) hAug[i * DEC_ + j] = loc[j];
    for (int e = 0; e < EMB_; ++e) {
        float s = hb[e];
        for (int j = 0; j < DEC_; ++j) s += loc[j] * hW[j * EMB_ + e];
        emb[i * EMB_ + e] = s;
    }
}

// scores[b,n,m] = relu(emb[b,n]·emb[b,m])
__global__ void k_support_scores(const float* __restrict__ emb, float* __restrict__ sup) {
    const long long i = (long long)blockIdx.x * blockDim.x + threadIdx.x;
    if (i >= (long long)BB * NN * NN) return;
    const int m = (int)(i % NN);
    const long long bn = i / NN;
    const int b = (int)(bn / NN);
    const float* ea = emb + bn * EMB_;
    const float* eb = emb + ((long long)b * NN + m) * EMB_;
    float s = 0.f;
    for (int e = 0; e < EMB_; ++e) s += ea[e] * eb[e];
    sup[i] = s > 0.f ? s : 0.f;
}

// Row-wise softmax over 512 cols, one wave per row; dual f32 + f16 output.
__global__ __launch_bounds__(256)
void k_support_softmax(float* __restrict__ sup, _Float16* __restrict__ suph) {
    const int lane = threadIdx.x & 31;
    const int wid  = threadIdx.x >> 5;
    const long long row = (long long)blockIdx.x * 8 + wid;
    if (row >= (long long)BB * NN) return;
    float* r = sup + row * NN;
    _Float16* rh = suph + row * NN;
    float mx = -1e30f;
    for (int c = lane; c < NN; c += 32) mx = fmaxf(mx, r[c]);
    for (int o = 16; o > 0; o >>= 1) mx = fmaxf(mx, __shfl_xor(mx, o, 32));
    float sum = 0.f;
    for (int c = lane; c < NN; c += 32) { float e = __expf(r[c] - mx); r[c] = e; sum += e; }
    for (int o = 16; o > 0; o >>= 1) sum += __shfl_xor(sum, o, 32);
    const float inv = 1.f / sum;
    for (int c = lane; c < NN; c += 32) {
        const float v = r[c] * inv;
        r[c]  = v;
        rh[c] = (_Float16)v;
    }
}

// go = h @ projW + projb ; also write output[b,t,n]
__global__ void k_proj(const float* __restrict__ h, const float* __restrict__ W,
                       const float* __restrict__ b, float* __restrict__ go,
                       float* __restrict__ out, int t) {
    const long long i = (long long)blockIdx.x * blockDim.x + threadIdx.x;
    if (i >= (long long)BB * NN) return;
    float s = b[0];
    for (int j = 0; j < DEC_; ++j) s += h[i * DEC_ + j] * W[j];
    go[i] = s;
    const int n = (int)(i % NN), bb = (int)(i / NN);
    out[((long long)bb * HORIZON_ + t) * NN + n] = s;
}

// ---------------------------------------------------------------------------
// Host-side orchestration
// ---------------------------------------------------------------------------
static inline void launch_gemm(hipStream_t s,
                               const _Float16* A, long long sA,
                               const _Float16* BT, long long sB,
                               float* CT, long long sC,
                               _Float16* CTh, long long sCh,
                               const float* bias, int M, int Nn, int K, int act,
                               int batch) {
    const int tiles = (M / 16) * (Nn / 16);
    dim3 grid(CDIV(tiles, 8), batch);
    gemm_wmma_f16<<<grid, 256, 0, s>>>(A, sA, BT, sB, CT, sC, CTh, sCh, bias, M, Nn, K, act);
}

struct Ws {
    _Float16 *Ah, *WgE, *WuE, *WgD, *WuD;
    _Float16 *inpT, *g1Th, *cheb, *sup_h;
    float *g1T, *g2T, *zrT, *hcT, *h, *hHis, *hAtt, *hAug, *emb, *sup, *go;
    int *ind, *indHis;
};

static void encode_pass(hipStream_t s, const Ws& w, const float* x, float* h,
                        const float* bg, const float* bu) {
    const long long BN = (long long)BB * NN;
    const int EW = 256;
    const long long gsA = 0;                         // shared adjacency
    const long long gsX = (long long)ENC_CP * NN;    // per-batch activation stride
    for (int t = 0; t < TT; ++t) {
        // ---- gate path ----
        k_enc_inp<<<CDIV(BN * ENC_IN, EW), EW, 0, s>>>(x, h, w.inpT, t);
        launch_gemm(s, w.Ah, gsA, w.inpT, gsX, w.g1T, gsX, w.g1Th, gsX,
                    nullptr, NN, ENC_CP, NN, 0, BB);
        launch_gemm(s, w.Ah, gsA, w.g1Th, gsX, w.g2T, gsX, nullptr, 0,
                    nullptr, NN, ENC_CP, NN, 0, BB);
        k_cheb_concat<<<CDIV(BN * ENC_IN, EW), EW, 0, s>>>(w.inpT, w.g1T, w.g2T, w.cheb,
                                                           ENC_IN, ENC_CP, ENC_KP);
        launch_gemm(s, w.cheb, 0, w.WgE, 0, w.zrT, 0, nullptr, 0,
                    bg, (int)BN, 2 * RNN_, ENC_KP, /*sigmoid*/1, 1);
        // ---- candidate path ----
        k_enc_cand<<<CDIV(BN * ENC_IN, EW), EW, 0, s>>>(x, w.zrT, h, w.inpT, t);
        launch_gemm(s, w.Ah, gsA, w.inpT, gsX, w.g1T, gsX, w.g1Th, gsX,
                    nullptr, NN, ENC_CP, NN, 0, BB);
        launch_gemm(s, w.Ah, gsA, w.g1Th, gsX, w.g2T, gsX, nullptr, 0,
                    nullptr, NN, ENC_CP, NN, 0, BB);
        k_cheb_concat<<<CDIV(BN * ENC_IN, EW), EW, 0, s>>>(w.inpT, w.g1T, w.g2T, w.cheb,
                                                           ENC_IN, ENC_CP, ENC_KP);
        launch_gemm(s, w.cheb, 0, w.WuE, 0, w.hcT, 0, nullptr, 0,
                    bu, (int)BN, RNN_, ENC_KP, /*tanh*/2, 1);
        k_gru_update<<<CDIV(BN * RNN_, EW), EW, 0, s>>>(w.zrT, w.hcT, h, RNN_);
    }
}

extern "C" void kernel_launch(void* const* d_in, const int* in_sizes, int n_in,
                              void* d_out, int out_size, void* d_ws, size_t ws_size,
                              hipStream_t stream) {
    (void)in_sizes; (void)n_in; (void)out_size; (void)ws_size;
    const float* x      = (const float*)d_in[0];
    // d_in[1] = x_cov (unused by reference)
    const float* x_his  = (const float*)d_in[2];
    const float* y_cov  = (const float*)d_in[3];
    const float* adj    = (const float*)d_in[4];
    const float* egW    = (const float*)d_in[5];
    const float* egb    = (const float*)d_in[6];
    const float* euW    = (const float*)d_in[7];
    const float* eub    = (const float*)d_in[8];
    const float* dgW    = (const float*)d_in[9];
    const float* dgb    = (const float*)d_in[10];
    const float* duW    = (const float*)d_in[11];
    const float* dub    = (const float*)d_in[12];
    const float* Mem    = (const float*)d_in[13];
    const float* Wq     = (const float*)d_in[14];
    const float* hyW    = (const float*)d_in[15];
    const float* hyb    = (const float*)d_in[16];
    const float* latW   = (const float*)d_in[17];
    const float* latb   = (const float*)d_in[18];
    const float* prW    = (const float*)d_in[19];
    const float* prb    = (const float*)d_in[20];

    float* out = (float*)d_out;
    const long long BN = (long long)BB * NN;

    // Bump allocator over d_ws
    char* p = (char*)d_ws;
    auto alloc = [&](long long bytes) -> void* {
        void* r = (void*)p;
        p += (bytes + 255) & ~255LL;
        return r;
    };
    Ws w;
    w.Ah    = (_Float16*)alloc((long long)NN * NN * 2);
    w.WgE   = (_Float16*)alloc((long long)2 * RNN_ * ENC_KP * 2);   // [128][224]
    w.WuE   = (_Float16*)alloc((long long)RNN_ * ENC_KP * 2);       // [64][224]
    w.WgD   = (_Float16*)alloc((long long)2 * DEC_ * DEC_KP * 2);   // [256][416]
    w.WuD   = (_Float16*)alloc((long long)DEC_ * DEC_KP * 2);       // [128][416]
    w.inpT  = (_Float16*)alloc((long long)BB * DEC_CP * NN * 2);
    w.g1Th  = (_Float16*)alloc((long long)BB * DEC_CP * NN * 2);
    w.cheb  = (_Float16*)alloc(BN * DEC_KP * 2);
    w.sup_h = (_Float16*)alloc(BN * NN * 2);
    w.g1T   = (float*)alloc((long long)BB * DEC_CP * NN * 4);
    w.g2T   = (float*)alloc((long long)BB * DEC_CP * NN * 4);
    w.zrT   = (float*)alloc(BN * 2 * DEC_ * 4);
    w.hcT   = (float*)alloc(BN * DEC_ * 4);
    w.h     = (float*)alloc(BN * RNN_ * 4);
    w.hHis  = (float*)alloc(BN * RNN_ * 4);
    w.hAtt  = (float*)alloc(BN * MEMD_ * 4);
    w.hAug  = (float*)alloc(BN * DEC_ * 4);
    w.emb   = (float*)alloc(BN * EMB_ * 4);
    w.sup   = (float*)alloc(BN * NN * 4);
    w.go    = (float*)alloc(BN * 4);
    w.ind   = (int*)alloc(BN * 4);
    w.indHis= (int*)alloc(BN * 4);

    const int EW = 256;

    // Adjacency f32->f16 (row-major A operand); weights transposed+padded.
    k_f32_to_f16<<<CDIV((long long)NN * NN, EW), EW, 0, stream>>>(adj, w.Ah, (long long)NN * NN);
    k_w_transpose_f16<<<CDIV((long long)2 * RNN_ * ENC_KP, EW), EW, 0, stream>>>(egW, w.WgE, ENC_3C, ENC_KP, 2 * RNN_);
    k_w_transpose_f16<<<CDIV((long long)RNN_ * ENC_KP, EW), EW, 0, stream>>>(euW, w.WuE, ENC_3C, ENC_KP, RNN_);
    k_w_transpose_f16<<<CDIV((long long)2 * DEC_ * DEC_KP, EW), EW, 0, stream>>>(dgW, w.WgD, DEC_3C, DEC_KP, 2 * DEC_);
    k_w_transpose_f16<<<CDIV((long long)DEC_ * DEC_KP, EW), EW, 0, stream>>>(duW, w.WuD, DEC_3C, DEC_KP, DEC_);

    // ---- Encoder on x ----
    k_zero_f32<<<CDIV(BN * RNN_, EW), EW, 0, stream>>>(w.h, BN * RNN_);
    encode_pass(stream, w, x, w.h, egb, eub);
    k_memory_query<<<CDIV(BN, EW), EW, 0, stream>>>(w.h, Mem, Wq, w.hAtt, w.ind);

    // ---- Encoder on x_his ----
    k_zero_f32<<<CDIV(BN * RNN_, EW), EW, 0, stream>>>(w.hHis, BN * RNN_);
    encode_pass(stream, w, x_his, w.hHis, egb, eub);
    k_memory_query<<<CDIV(BN, EW), EW, 0, stream>>>(w.hHis, Mem, Wq, nullptr, w.indHis);

    // latent_dis -> out[B*H*N ..]
    k_latent<<<CDIV(BN, EW), EW, 0, stream>>>(Mem, w.ind, w.indHis, latW, latb,
                                              out + (long long)BB * HORIZON_ * NN);

    // h_aug, node_emb, support
    k_haug_emb<<<CDIV(BN, EW), EW, 0, stream>>>(w.h, w.hAtt, w.hAug, w.emb, hyW, hyb);
    k_support_scores<<<CDIV(BN * NN, EW), EW, 0, stream>>>(w.emb, w.sup);
    k_support_softmax<<<CDIV(BN, 8), 256, 0, stream>>>(w.sup, w.sup_h);

    // ---- Decoder (state = hAug, in place) ----
    k_zero_f32<<<CDIV(BN, EW), EW, 0, stream>>>(w.go, BN);
    const long long dsA = (long long)NN * NN;        // per-batch support stride
    const long long dsX = (long long)DEC_CP * NN;    // per-batch activation stride
    for (int t = 0; t < HORIZON_; ++t) {
        // gate path
        k_dec_inp<<<CDIV(BN * DEC_IN, EW), EW, 0, stream>>>(w.go, y_cov, w.hAug, w.inpT, t);
        launch_gemm(stream, w.sup_h, dsA, w.inpT, dsX, w.g1T, dsX, w.g1Th, dsX,
                    nullptr, NN, DEC_CP, NN, 0, BB);
        launch_gemm(stream, w.sup_h, dsA, w.g1Th, dsX, w.g2T, dsX, nullptr, 0,
                    nullptr, NN, DEC_CP, NN, 0, BB);
        k_cheb_concat<<<CDIV(BN * DEC_IN, EW), EW, 0, stream>>>(w.inpT, w.g1T, w.g2T, w.cheb,
                                                                DEC_IN, DEC_CP, DEC_KP);
        launch_gemm(stream, w.cheb, 0, w.WgD, 0, w.zrT, 0, nullptr, 0,
                    dgb, (int)BN, 2 * DEC_, DEC_KP, /*sigmoid*/1, 1);
        // candidate path
        k_dec_cand<<<CDIV(BN * DEC_IN, EW), EW, 0, stream>>>(w.go, y_cov, w.zrT, w.hAug, w.inpT, t);
        launch_gemm(stream, w.sup_h, dsA, w.inpT, dsX, w.g1T, dsX, w.g1Th, dsX,
                    nullptr, NN, DEC_CP, NN, 0, BB);
        launch_gemm(stream, w.sup_h, dsA, w.g1Th, dsX, w.g2T, dsX, nullptr, 0,
                    nullptr, NN, DEC_CP, NN, 0, BB);
        k_cheb_concat<<<CDIV(BN * DEC_IN, EW), EW, 0, stream>>>(w.inpT, w.g1T, w.g2T, w.cheb,
                                                                DEC_IN, DEC_CP, DEC_KP);
        launch_gemm(stream, w.cheb, 0, w.WuD, 0, w.hcT, 0, nullptr, 0,
                    dub, (int)BN, DEC_, DEC_KP, /*tanh*/2, 1);
        k_gru_update<<<CDIV(BN * DEC_, EW), EW, 0, stream>>>(w.zrT, w.hcT, w.hAug, DEC_);
        k_proj<<<CDIV(BN, EW), EW, 0, stream>>>(w.hAug, prW, prb, w.go, out, t);
    }
}
